// TernaryInvertedResidual_31164282699833
// MI455X (gfx1250) — matrix operand order repaired and checked by reading, not compiled, not run on hardware
//
#include <hip/hip_runtime.h>
#include <hip/hip_bf16.h>

typedef __attribute__((ext_vector_type(16))) _Float16 v16h;
typedef __attribute__((ext_vector_type(8)))  _Float16 v8h;
typedef __attribute__((ext_vector_type(8)))  float    v8f;
typedef __attribute__((ext_vector_type(4)))  unsigned u32x4;
typedef __attribute__((ext_vector_type(8)))  int      i32x8;
typedef __attribute__((ext_vector_type(4)))  int      i32x4;

#if defined(__has_builtin)
#if __has_builtin(__builtin_amdgcn_tensor_load_to_lds) && \
    __has_builtin(__builtin_amdgcn_s_wait_tensorcnt)
#define USE_TDM 1
#endif
#endif
#ifndef USE_TDM
#define USE_TDM 0
#endif

#define EPS 1e-5f
#define HW  12544   // 112*112
#define WID 112
#define IMGS 16

__device__ inline v8f wmma_f16(v16h a, v16h b, v8f c) {
  return __builtin_amdgcn_wmma_f32_16x16x32_f16(
      false, a, false, b, (short)0, c, false, false);
}

// Build a v16h fragment from two 16-byte chunks.
__device__ inline v16h frag2(const _Float16* p0, const _Float16* p1) {
  v8h x0 = *(const v8h*)p0;
  v8h x1 = *(const v8h*)p1;
  v16h r;
#pragma unroll
  for (int i = 0; i < 8; i++) { r[i] = x0[i]; r[8 + i] = x1[i]; }
  return r;
}

// A fragment, row-major [row][32] tile.
// Lane l: row M=(l&15); halves 0..7 = K 8*hi..,
// halves 8..15 = K 16+8*hi.. (ISA 16-bit A layout).
__device__ inline v16h load_A32(const _Float16* lds, int row, int hi) {
  const _Float16* b = lds + row * 32;
  return frag2(b + 8 * hi, b + 16 + 8 * hi);
}
// A fragment from padded [row][200] tile (conv3 weights).
__device__ inline v16h load_A200(const _Float16* lds, int row, int koff, int hi) {
  const _Float16* b = lds + row * 200 + koff;
  return frag2(b + 8 * hi, b + 16 + 8 * hi);
}
// B fragment from [n][k] tile, k contiguous with per-column stride.
// Lane l: col N=(l&15); halves 0..7 = K 8*hi.., halves 8..15 = K 16+8*hi.. (matches A).
__device__ inline v16h load_B(const _Float16* lds, int col, int stride, int koff,
                              int hi) {
  const _Float16* b = lds + col * stride + koff;
  return frag2(b + 8 * hi, b + 16 + 8 * hi);
}

__device__ inline v8h zero8h() {
  v8h z;
#pragma unroll
  for (int i = 0; i < 8; i++) z[i] = (_Float16)0.f;
  return z;
}

// ---------------------------------------------------------------------------
// Ternarize + BN-fold prep (single block; negligible cost).
// mode 0: w1 [192][32] -> Wq[co][ci]
// mode 1: w2 [192][192][3][3] -> Wq[ky*3+kx][co][ci]
// mode 2: w3 [32][192] -> Wq[co][ci]
// ---------------------------------------------------------------------------
__global__ __launch_bounds__(256) void k_prep(
    const float* __restrict__ w, int nelem,
    const float* __restrict__ g, const float* __restrict__ b,
    const float* __restrict__ m, const float* __restrict__ v, int C,
    _Float16* __restrict__ Wq, float* __restrict__ scale,
    float* __restrict__ shift, int mode) {
  __shared__ float red[256];
  __shared__ float red2[256];
  __shared__ float s_delta, s_alpha;
  const int t = threadIdx.x;

  float s = 0.f;
  for (int i = t; i < nelem; i += 256) s += fabsf(w[i]);
  red[t] = s;
  __syncthreads();
  for (int o = 128; o > 0; o >>= 1) {
    if (t < o) red[t] += red[t + o];
    __syncthreads();
  }
  if (t == 0) s_delta = 0.7f * red[0] / (float)nelem;
  __syncthreads();
  const float delta = s_delta;

  float ms = 0.f, cnt = 0.f;
  for (int i = t; i < nelem; i += 256) {
    float a = fabsf(w[i]);
    if (a > delta) { ms += a; cnt += 1.f; }
  }
  red[t] = ms;
  red2[t] = cnt;
  __syncthreads();
  for (int o = 128; o > 0; o >>= 1) {
    if (t < o) { red[t] += red[t + o]; red2[t] += red2[t + o]; }
    __syncthreads();
  }
  if (t == 0) s_alpha = red[0] / fmaxf(red2[0], 1.f);
  __syncthreads();
  const float alpha = s_alpha;

  for (int i = t; i < nelem; i += 256) {
    float wv = w[i];
    float sg = (wv > 0.f) ? 1.f : ((wv < 0.f) ? -1.f : 0.f);
    float q = (fabsf(wv) > delta) ? sg : 0.f;
    int o;
    if (mode == 1) {
      int co = i / 1728;
      int rem = i - co * 1728;
      int ci = rem / 9;
      int kk = rem - ci * 9;
      o = (kk * 192 + co) * 192 + ci;
    } else {
      o = i;
    }
    Wq[o] = (_Float16)q;
  }

  for (int c = t; c < C; c += 256) {
    float s0 = g[c] * rsqrtf(v[c] + EPS);
    scale[c] = alpha * s0;
    shift[c] = b[c] - m[c] * s0;
  }
}

// ---------------------------------------------------------------------------
// conv1: 1x1 expand. GEMM M=192 (tile 64), K=32, N tile 128.
// x NCHW fp32 -> f16 on staging. Output h1 in NHWC f16: [img][pixel][192].
// grid: (16*98, 3)
// ---------------------------------------------------------------------------
__global__ __launch_bounds__(256) void k_conv1(
    const float* __restrict__ x, const _Float16* __restrict__ Wq,
    const float* __restrict__ scale, const float* __restrict__ shift,
    _Float16* __restrict__ h1) {
  __shared__ _Float16 Alds[64 * 32];
  __shared__ _Float16 Blds[128 * 32];
  const int t = threadIdx.x;
  const int ntile = blockIdx.x;
  const int mblk = blockIdx.y;
  const int img = ntile / 98;
  const int q0 = (ntile - img * 98) * 128;
  const float* xin = x + (size_t)img * 32 * HW;

  {  // stage A: 64x32 ternary weights
    int mm = t >> 2;
    int kk = (t & 3) * 8;
    *(v8h*)(Alds + mm * 32 + kk) =
        *(const v8h*)(Wq + (size_t)(mblk * 64 + mm) * 32 + kk);
  }
  {  // stage B transposed [n][k], fp32 -> f16
    int bn = t & 127;
    int kb = (t >> 7) * 16;
    v8h t0, t1;
#pragma unroll
    for (int kk = 0; kk < 8; kk++)
      t0[kk] = (_Float16)xin[(size_t)(kb + kk) * HW + q0 + bn];
#pragma unroll
    for (int kk = 0; kk < 8; kk++)
      t1[kk] = (_Float16)xin[(size_t)(kb + 8 + kk) * HW + q0 + bn];
    *(v8h*)(Blds + bn * 32 + kb) = t0;
    *(v8h*)(Blds + bn * 32 + kb + 8) = t1;
  }
  __syncthreads();

  const int lane = t & 31, wid = t >> 5;
  const int mw = wid >> 1, nw = wid & 1;
  const int hi = (lane >> 4) & 1, ll = lane & 15;

  v16h a = load_A32(Alds, mw * 16 + ll, hi);
  v8f z = {0.f, 0.f, 0.f, 0.f, 0.f, 0.f, 0.f, 0.f};
  v8f c[4] = {z, z, z, z};
#pragma unroll
  for (int tt = 0; tt < 4; tt++) {
    v16h bf = load_B(Blds, nw * 64 + tt * 16 + ll, 32, 0, hi);
    c[tt] = wmma_f16(a, bf, c[tt]);
  }

  // epilogue: BN + ReLU6 -> NHWC f16, vectorized b128 stores
  const int ch0 = mblk * 64 + mw * 16 + hi * 8;
#pragma unroll
  for (int tt = 0; tt < 4; tt++) {
    int colg = q0 + nw * 64 + tt * 16 + ll;
    v8h hv;
#pragma unroll
    for (int r = 0; r < 8; r++) {
      float y = c[tt][r] * scale[ch0 + r] + shift[ch0 + r];
      y = fminf(fmaxf(y, 0.f), 6.f);
      hv[r] = (_Float16)y;
    }
    *(v8h*)(h1 + ((size_t)img * HW + colg) * 192 + ch0) = hv;
  }
}

// ---------------------------------------------------------------------------
// Fused conv2 (3x3, pad 1) + conv3 (1x1 project) + residual.
// Block: 256 threads = 8 waves, N tile = 128 pixels, all 192 conv2 channels.
// conv2: waves (mgrp 0..3 x 48 rows, ngrp 0..1 x 64 cols), 12 WMMA tiles/wave.
// Double-buffered LDS staging: weight tiles via Tensor Data Mover (wave 0
// issues one tensor_load_to_lds per K-step, tracked by TENSORcnt), activation
// tiles via register-pipelined global loads (per-pixel zero padding).
// Single barrier per K-step. conv2 result (BN+ReLU6, f16) staged in LDS
// [pixel][200], then conv3 GEMM M=32,K=192 from LDS, + BN + residual -> fp32.
// grid: 16*98 blocks
// ---------------------------------------------------------------------------
__global__ __launch_bounds__(256) void k_conv23(
    const _Float16* __restrict__ h1,   // NHWC [img][HW][192]
    const _Float16* __restrict__ W2q,  // [9][192][192]
    const float* __restrict__ sc2, const float* __restrict__ sh2,
    const _Float16* __restrict__ W3q,  // [32][192]
    const float* __restrict__ sc3, const float* __restrict__ sh3,
    const float* __restrict__ xres, float* __restrict__ out) {
  __shared__ _Float16 Alds[2 * 6144];   // conv2 A dbl-buf; later W3 [32][200]
  __shared__ _Float16 Blds[2 * 4096];   // conv2 B dbl-buf [n][32]
  __shared__ _Float16 Hlds[128 * 200];  // conv2 output tile [n][ch] (pad 200)
  const int t = threadIdx.x;
  const int ntile = blockIdx.x;
  const int img = ntile / 98;
  const int q0 = (ntile - img * 98) * 128;
  const _Float16* hin = h1 + (size_t)img * HW * 192;

  const int lane = t & 31, wid = t >> 5;
  const int mgrp = wid >> 1, ngrp = wid & 1;
  const int hi = (lane >> 4) & 1, ll = lane & 15;

  // B staging: one pixel per 2 threads
  const int bn = t >> 1;        // 0..127
  const int kq = (t & 1) * 16;  // 0 or 16
  const int p = q0 + bn;
  const int phh = p / WID;
  const int pww = p - phh * WID;

  // A staging (manual fallback): thread covers rows aco, aco+64, aco+128
  const int aco = t >> 2;
  const int aci = (t & 3) * 8;

  auto issueB = [&](int it, v8h& b0, v8h& b1) {
    int tap = it / 6;
    int ks = it - tap * 6;
    int ky = tap / 3, kx = tap - ky * 3;
    int hh = phh + ky - 1, ww = pww + kx - 1;
    bool ok = (hh >= 0) & (hh < WID) & (ww >= 0) & (ww < WID);
    if (ok) {
      const _Float16* bsrc = hin + (size_t)(hh * WID + ww) * 192 + ks * 32 + kq;
      b0 = *(const v8h*)(bsrc);
      b1 = *(const v8h*)(bsrc + 8);
    } else {
      b0 = zero8h();
      b1 = zero8h();
    }
  };
  auto commitB = [&](int buf, v8h b0, v8h b1) {
    _Float16* B = Blds + buf * 4096;
    *(v8h*)(B + bn * 32 + kq) = b0;
    *(v8h*)(B + bn * 32 + kq + 8) = b1;
  };

#if USE_TDM
  // TDM: copy the 192(row) x 32(half) weight tile for K-step `it` into
  // Alds[buf] as contiguous [co][32] (tile scan order X=32 elems, Y=192 rows).
  auto issueA_tdm = [&](int it, int buf) {
    int tap = it / 6;
    int ks = it - tap * 6;
    unsigned lds_addr = (unsigned)(size_t)(const void*)(Alds + buf * 6144);
    unsigned long long ga =
        (unsigned long long)(size_t)(W2q + (size_t)tap * 192 * 192 + ks * 32);
    u32x4 g0;
    g0[0] = 1u;                                   // count=1 (user descriptor)
    g0[1] = lds_addr;                             // lds_addr
    g0[2] = (unsigned)(ga & 0xffffffffu);         // global_addr[31:0]
    g0[3] = (unsigned)((ga >> 32) & 0x1ffffffu)   // global_addr[56:32]
            | (2u << 30);                         // type=2 (image)
    i32x8 g1;
    g1[0] = 1 << 16;        // workgroup_mask=0, data_size=1 (2 bytes)
    g1[1] = 32 << 16;       // tensor_dim0[15:0] = 32
    g1[2] = 192 << 16;      // tensor_dim0 hi=0 | tensor_dim1[15:0] = 192
    g1[3] = 32 << 16;       // tensor_dim1 hi=0 | tile_dim0 = 32
    g1[4] = 192;            // tile_dim1 = 192, tile_dim2 = 0
    g1[5] = 192;            // tensor_dim0_stride[31:0] = 192 elements
    g1[6] = 0;              // stride0 hi | tensor_dim1_stride lo
    g1[7] = 0;
    i32x4 zz4 = {0, 0, 0, 0};
    i32x8 zz8 = {0, 0, 0, 0, 0, 0, 0, 0};
    __builtin_amdgcn_tensor_load_to_lds(g0, g1, zz4, zz4, zz8, 0);
  };
#else
  auto issueA = [&](int it, v8h& a0, v8h& a1, v8h& a2) {
    int tap = it / 6;
    int ks = it - tap * 6;
    const _Float16* asrc =
        W2q + ((size_t)tap * 192 + aco) * 192 + ks * 32 + aci;
    a0 = *(const v8h*)(asrc);
    a1 = *(const v8h*)(asrc + (size_t)64 * 192);
    a2 = *(const v8h*)(asrc + (size_t)128 * 192);
  };
  auto commitA = [&](int buf, v8h a0, v8h a1, v8h a2) {
    _Float16* A = Alds + buf * 6144;
    *(v8h*)(A + aco * 32 + aci) = a0;
    *(v8h*)(A + (aco + 64) * 32 + aci) = a1;
    *(v8h*)(A + (aco + 128) * 32 + aci) = a2;
  };
#endif

  v8f z = {0.f, 0.f, 0.f, 0.f, 0.f, 0.f, 0.f, 0.f};
  v8f c2[12];  // [mt][nt] flattened mt*4+nt
#pragma unroll
  for (int i = 0; i < 12; i++) c2[i] = z;

  {  // prologue: fill buffer 0
    v8h b0, b1;
    issueB(0, b0, b1);
#if USE_TDM
    if (wid == 0) issueA_tdm(0, 0);
    commitB(0, b0, b1);
    if (wid == 0) __builtin_amdgcn_s_wait_tensorcnt(0);
#else
    v8h a0, a1, a2;
    issueA(0, a0, a1, a2);
    commitA(0, a0, a1, a2);
    commitB(0, b0, b1);
#endif
  }
  __syncthreads();

  for (int it = 0; it < 54; it++) {
    const int buf = it & 1;
    v8h nb0, nb1;
#if USE_TDM
    if (it + 1 < 54) {
      if (wid == 0) issueA_tdm(it + 1, buf ^ 1);
      issueB(it + 1, nb0, nb1);
    }
#else
    v8h na0, na1, na2;
    if (it + 1 < 54) {
      issueA(it + 1, na0, na1, na2);
      issueB(it + 1, nb0, nb1);
    }
#endif

    const _Float16* A = Alds + buf * 6144;
    const _Float16* B = Blds + buf * 4096;
    v16h a[3];
#pragma unroll
    for (int mt = 0; mt < 3; mt++)
      a[mt] = load_A32(A, mgrp * 48 + mt * 16 + ll, hi);
#pragma unroll
    for (int nt = 0; nt < 4; nt++) {
      v16h bf = load_B(B, ngrp * 64 + nt * 16 + ll, 32, 0, hi);
#pragma unroll
      for (int mt = 0; mt < 3; mt++)
        c2[mt * 4 + nt] = wmma_f16(a[mt], bf, c2[mt * 4 + nt]);
    }

    if (it + 1 < 54) {
      commitB(buf ^ 1, nb0, nb1);
#if USE_TDM
      if (wid == 0) __builtin_amdgcn_s_wait_tensorcnt(0);
#else
      commitA(buf ^ 1, na0, na1, na2);
#endif
    }
    __syncthreads();
  }

  // conv2 epilogue: BN + ReLU6 -> Hlds [n][ch] (f16), vectorized v8h stores
#pragma unroll
  for (int mt = 0; mt < 3; mt++) {
    const int ch0 = mgrp * 48 + mt * 16 + hi * 8;
#pragma unroll
    for (int nt = 0; nt < 4; nt++) {
      const int n = ngrp * 64 + nt * 16 + ll;
      v8h hv;
#pragma unroll
      for (int r = 0; r < 8; r++) {
        float y = c2[mt * 4 + nt][r] * sc2[ch0 + r] + sh2[ch0 + r];
        y = fminf(fmaxf(y, 0.f), 6.f);
        hv[r] = (_Float16)y;
      }
      *(v8h*)(Hlds + n * 200 + ch0) = hv;
    }
  }
  // stage W3 [32][192] into (now free) Alds with padded row stride 200
  for (int i = t; i < 768; i += 256) {
    int co = i / 24;
    int c8 = (i - co * 24) * 8;
    *(v8h*)(Alds + co * 200 + c8) = *(const v8h*)(W3q + (size_t)co * 192 + c8);
  }
  __syncthreads();

  // conv3: M=32, K=192 (6 steps), each wave owns one 16-col slice
  const int ntw = wid;  // 0..7
  v8f c3[2] = {z, z};
  for (int ks = 0; ks < 6; ks++) {
#pragma unroll
    for (int mt = 0; mt < 2; mt++) {
      v16h a = load_A200(Alds, mt * 16 + ll, ks * 32, hi);
      v16h bf = load_B(Hlds, ntw * 16 + ll, 200, ks * 32, hi);
      c3[mt] = wmma_f16(a, bf, c3[mt]);
    }
  }

  // conv3 epilogue: BN + residual, fp32 out (NCHW)
#pragma unroll
  for (int mt = 0; mt < 2; mt++) {
    const int colg = q0 + ntw * 16 + ll;
#pragma unroll
    for (int r = 0; r < 8; r++) {
      int ch = mt * 16 + hi * 8 + r;
      size_t idx = ((size_t)img * 32 + ch) * HW + colg;
      out[idx] = xres[idx] + (c3[mt][r] * sc3[ch] + sh3[ch]);
    }
  }
}

// ---------------------------------------------------------------------------
extern "C" void kernel_launch(void* const* d_in, const int* in_sizes, int n_in,
                              void* d_out, int out_size, void* d_ws,
                              size_t ws_size, hipStream_t stream) {
  const float* x  = (const float*)d_in[0];
  const float* w1 = (const float*)d_in[1];
  const float* g1 = (const float*)d_in[2];
  const float* b1 = (const float*)d_in[3];
  const float* m1 = (const float*)d_in[4];
  const float* v1 = (const float*)d_in[5];
  const float* w2 = (const float*)d_in[6];
  const float* g2 = (const float*)d_in[7];
  const float* b2 = (const float*)d_in[8];
  const float* m2 = (const float*)d_in[9];
  const float* v2 = (const float*)d_in[10];
  const float* w3 = (const float*)d_in[11];
  const float* g3 = (const float*)d_in[12];
  const float* b3 = (const float*)d_in[13];
  const float* m3 = (const float*)d_in[14];
  const float* v3 = (const float*)d_in[15];
  float* out = (float*)d_out;

  char* ws = (char*)d_ws;
  size_t off = 0;
  auto take = [&](size_t bytes) {
    char* p = ws + off;
    off = (off + bytes + 255) & ~(size_t)255;
    return p;
  };
  _Float16* W1q = (_Float16*)take(192 * 32 * sizeof(_Float16));
  _Float16* W2q = (_Float16*)take((size_t)9 * 192 * 192 * sizeof(_Float16));
  _Float16* W3q = (_Float16*)take(32 * 192 * sizeof(_Float16));
  float* sc1 = (float*)take(192 * sizeof(float));
  float* sh1 = (float*)take(192 * sizeof(float));
  float* sc2 = (float*)take(192 * sizeof(float));
  float* sh2 = (float*)take(192 * sizeof(float));
  float* sc3 = (float*)take(32 * sizeof(float));
  float* sh3 = (float*)take(32 * sizeof(float));
  const size_t hsz = (size_t)IMGS * HW * 192;  // NHWC
  _Float16* h1 = (_Float16*)take(hsz * sizeof(_Float16));

  k_prep<<<1, 256, 0, stream>>>(w1, 192 * 32, g1, b1, m1, v1, 192, W1q, sc1,
                                sh1, 0);
  k_prep<<<1, 256, 0, stream>>>(w2, 192 * 192 * 9, g2, b2, m2, v2, 192, W2q,
                                sc2, sh2, 1);
  k_prep<<<1, 256, 0, stream>>>(w3, 32 * 192, g3, b3, m3, v3, 32, W3q, sc3,
                                sh3, 2);

  dim3 grid1(IMGS * 98, 3);
  k_conv1<<<grid1, 256, 0, stream>>>(x, W1q, sc1, sh1, h1);
  k_conv23<<<dim3(IMGS * 98), 256, 0, stream>>>(h1, W2q, sc2, sh2, W3q, sc3,
                                                sh3, x, out);
}
